// Siamese_83760452206708
// MI455X (gfx1250) — matrix operand order repaired and checked by reading, not compiled, not run on hardware
//
#include <hip/hip_runtime.h>

typedef __attribute__((ext_vector_type(16))) _Float16 v16h;
typedef __attribute__((ext_vector_type(8)))  _Float16 v8h;
typedef __attribute__((ext_vector_type(8)))  float    v8f;

#define NB   32                 // batch
#define CH   32                 // channels
#define RES  96
#define SQ   64                 // queries per image
#define PIX  (RES*RES)          // 9216
#define PR   (RES+2)            // 98 (zero-padded)
#define KDIM 288                // 9*32 = (u,v,c)
#define LDS_STRIDE 296          // padded K stride for LDS A tile

// ---------------------------------------------------------------------------
// Pass 1a: fBn[n][y][x][c] = fB / sum_c(fB^2), zero-padded, f16 channel-last.
// Channel-last layout lets each WMMA lane fetch its 16 contiguous K-halves
// (K = c within a fixed (u,v) group) with two global_load_b128.
// ---------------------------------------------------------------------------
__global__ void norm_fB_kernel(const float* __restrict__ fB,
                               _Float16* __restrict__ fBn) {
  int t = blockIdx.x * blockDim.x + threadIdx.x;
  const int total = NB * PR * PR;
  if (t >= total) return;
  int n   = t / (PR * PR);
  int rem = t % (PR * PR);
  int y = rem / PR, x = rem % PR;
  _Float16* dst = fBn + (size_t)t * CH;
  v8h* d = (v8h*)dst;
  if (y == 0 || y == PR - 1 || x == 0 || x == PR - 1) {
    v8h z = {};
    d[0] = z; d[1] = z; d[2] = z; d[3] = z;
    return;
  }
  int sy = y - 1, sx = x - 1;
  const float* src = fB + (size_t)n * CH * PIX + (size_t)sy * RES + sx;
  float vals[CH];
  float ssq = 0.f;
  #pragma unroll
  for (int c = 0; c < CH; ++c) {
    float v = src[(size_t)c * PIX];
    vals[c] = v;
    ssq += v * v;
  }
  float inv = (ssq > 0.f) ? (1.f / ssq) : 0.f;
  _Float16 h[CH];
  #pragma unroll
  for (int c = 0; c < CH; ++c) h[c] = (_Float16)(vals[c] * inv);
  const v8h* s = (const v8h*)h;
  d[0] = s[0]; d[1] = s[1]; d[2] = s[2]; d[3] = s[3];
}

// ---------------------------------------------------------------------------
// Pass 1b: A[n][s][k] f16, k = (u*3+v)*32 + c, with fA normalized by the
// per-pixel channel sum of squares. One wave per (query, uv); lane = channel.
// ---------------------------------------------------------------------------
__global__ void build_A_kernel(const float* __restrict__ fA,
                               const long long* __restrict__ q,
                               _Float16* __restrict__ A_ws) {
  int bid = blockIdx.x;                 // n*SQ + s
  int uv  = threadIdx.x >> 5;           // 0..8  (one wave each)
  int c   = threadIdx.x & 31;
  int n = bid / SQ;
  long long qy = q[(size_t)bid * 2 + 0];
  long long qx = q[(size_t)bid * 2 + 1];
  bool valid = (qy >= 0) && (qy < RES) && (qx >= 0) && (qx < RES);
  int iy = valid ? (int)qy : 0;         // clamp for safe addressing; output
  int ix = valid ? (int)qx : 0;         // is zeroed later if invalid
  int u = uv / 3, v = uv % 3;
  int py = iy + u - 1, px = ix + v - 1;
  bool inb = (py >= 0) && (py < RES) && (px >= 0) && (px < RES);
  float val = inb ? fA[(((size_t)n * CH + c) * RES + py) * RES + px] : 0.f;
  float ss = val * val;
  #pragma unroll
  for (int m = 16; m >= 1; m >>= 1) ss += __shfl_xor(ss, m, 32);
  float a = (ss > 0.f) ? (val / ss) : 0.f;
  A_ws[(size_t)bid * KDIM + uv * 32 + c] = (_Float16)a;
}

// ---------------------------------------------------------------------------
// Pass 2: corr[n] = A[n](64x288) @ B[n](288x9216) via v_wmma_f32_16x16x32_f16.
// Block = (pixel tile of 128, image n); 8 waves, each owns 16 pixels and all
// 4 M-tiles so every B fragment feeds 4 WMMAs. Writes corr into d_out.
// ---------------------------------------------------------------------------
__global__ void __launch_bounds__(256) corr_wmma_kernel(
    const _Float16* __restrict__ A_ws,
    const _Float16* __restrict__ fBn,
    float* __restrict__ out) {
  __shared__ _Float16 Alds[SQ * LDS_STRIDE];

  int n   = blockIdx.y;
  int tid = threadIdx.x;

  // Cooperative stage of the image's A matrix into LDS (dword granularity).
  const unsigned int* src = (const unsigned int*)(A_ws + (size_t)n * SQ * KDIM);
  for (int t = tid; t < SQ * (KDIM / 2); t += 256) {
    int row = t / (KDIM / 2), dw = t % (KDIM / 2);
    *(unsigned int*)&Alds[row * LDS_STRIDE + dw * 2] = src[row * (KDIM / 2) + dw];
  }
  __syncthreads();

  int wave = tid >> 5, lane = tid & 31;
  int g  = lane >> 4;                       // half-wave selects K sub-range
  int nl = lane & 15;                       // N within tile
  int p  = blockIdx.x * 128 + wave * 16 + nl;
  int i  = p / RES, j = p % RES;

  v8f acc0 = {}, acc1 = {}, acc2 = {}, acc3 = {};

  #pragma unroll
  for (int ks = 0; ks < 9; ++ks) {          // K-step = one (u,v) tap, 32 channels
    int u = ks / 3, v = ks % 3;
    // B fragment: lanes 0-15 hold K(=c)=0..15, lanes 16-31 hold K=16..31,
    // contiguous in channel-last fBn -> two b128 loads.
    const v8h* pb = (const v8h*)(fBn + (((size_t)n * PR + (i + u)) * PR + (j + v)) * CH);
    v8h blo = pb[2 * g], bhi = pb[2 * g + 1];
    v16h b = __builtin_shufflevector(blo, bhi,
             0,1,2,3,4,5,6,7,8,9,10,11,12,13,14,15);
    #pragma unroll
    for (int mt = 0; mt < 4; ++mt) {
      // A fragment: element e -> K=e+16g (matches B).
      int row = mt * 16 + nl;
      const v8h* pa = (const v8h*)&Alds[row * LDS_STRIDE + ks * 32];
      v8h alo = pa[2 * g], ahi = pa[2 * g + 1];
      v16h a = __builtin_shufflevector(alo, ahi,
               0,1,2,3,4,5,6,7,8,9,10,11,12,13,14,15);
      v8f& acc = (mt == 0) ? acc0 : (mt == 1) ? acc1 : (mt == 2) ? acc2 : acc3;
      acc = __builtin_amdgcn_wmma_f32_16x16x32_f16(
          /*neg_a=*/false, a, /*neg_b=*/false, b,
          /*c_mod=*/(short)0, acc, /*reuse_a=*/false, /*reuse_b=*/false);
    }
  }

  // D layout: VGPR r -> M = r + 8*g, N = lane&15.
  float* obase = out + (size_t)n * SQ * PIX + p;
  #pragma unroll
  for (int r = 0; r < 8; ++r) {
    int m = r + 8 * g;
    obase[(size_t)( 0 + m) * PIX] = acc0[r];
    obase[(size_t)(16 + m) * PIX] = acc1[r];
    obase[(size_t)(32 + m) * PIX] = acc2[r];
    obase[(size_t)(48 + m) * PIX] = acc3[r];
  }
}

// ---------------------------------------------------------------------------
// Pass 3: in-place softmax over the 9216 pixels of each (n,s) row; zero
// invalid queries. One 256-thread block per row (36 elements/thread).
// ---------------------------------------------------------------------------
__global__ void softmax_kernel(float* __restrict__ out,
                               const long long* __restrict__ q) {
  __shared__ float red[8];
  int bid = blockIdx.x;                     // n*SQ + s
  long long qy = q[(size_t)bid * 2], qx = q[(size_t)bid * 2 + 1];
  bool valid = (qy >= 0) && (qy < RES) && (qx >= 0) && (qx < RES);
  float* row = out + (size_t)bid * PIX;
  int tid = threadIdx.x;
  if (!valid) {
    #pragma unroll
    for (int k = 0; k < 36; ++k) row[tid + k * 256] = 0.f;
    return;
  }
  float vals[36];
  float mx = -3.4e38f;
  #pragma unroll
  for (int k = 0; k < 36; ++k) {
    float v = row[tid + k * 256];
    vals[k] = v;
    mx = fmaxf(mx, v);
  }
  int w = tid >> 5, lane = tid & 31;
  #pragma unroll
  for (int m = 16; m >= 1; m >>= 1) mx = fmaxf(mx, __shfl_xor(mx, m, 32));
  if (lane == 0) red[w] = mx;
  __syncthreads();
  if (tid < 32) {
    float x = (lane < 8) ? red[lane] : -3.4e38f;
    #pragma unroll
    for (int m = 4; m >= 1; m >>= 1) x = fmaxf(x, __shfl_xor(x, m, 32));
    if (lane == 0) red[0] = x;
  }
  __syncthreads();
  mx = red[0];
  __syncthreads();
  float lsum = 0.f;
  #pragma unroll
  for (int k = 0; k < 36; ++k) {
    vals[k] = expf(vals[k] - mx);
    lsum += vals[k];
  }
  #pragma unroll
  for (int m = 16; m >= 1; m >>= 1) lsum += __shfl_xor(lsum, m, 32);
  if (lane == 0) red[w] = lsum;
  __syncthreads();
  if (tid < 32) {
    float x = (lane < 8) ? red[lane] : 0.f;
    #pragma unroll
    for (int m = 4; m >= 1; m >>= 1) x += __shfl_xor(x, m, 32);
    if (lane == 0) red[0] = x;
  }
  __syncthreads();
  float inv = 1.f / red[0];
  #pragma unroll
  for (int k = 0; k < 36; ++k) row[tid + k * 256] = vals[k] * inv;
}

// ---------------------------------------------------------------------------
extern "C" void kernel_launch(void* const* d_in, const int* in_sizes, int n_in,
                              void* d_out, int out_size, void* d_ws, size_t ws_size,
                              hipStream_t stream) {
  (void)in_sizes; (void)n_in; (void)out_size; (void)ws_size;
  const float*     fA = (const float*)d_in[0];
  const float*     fB = (const float*)d_in[1];
  const long long* q  = (const long long*)d_in[2];   // int64 queries
  float* out = (float*)d_out;

  _Float16* A_ws = (_Float16*)d_ws;                          // 32*64*288 f16
  size_t aBytes = (size_t)NB * SQ * KDIM * sizeof(_Float16); // 1,179,648 B (16B aligned)
  _Float16* fBn = (_Float16*)((char*)d_ws + aBytes);         // 32*98*98*32 f16

  int totPix = NB * PR * PR;
  norm_fB_kernel<<<(totPix + 255) / 256, 256, 0, stream>>>(fB, fBn);
  build_A_kernel<<<NB * SQ, 288, 0, stream>>>(fA, q, A_ws);
  corr_wmma_kernel<<<dim3(PIX / 128, NB), 256, 0, stream>>>(A_ws, fBn, out);
  softmax_kernel<<<NB * SQ, 256, 0, stream>>>(out, q);
}
